// MultiHeadAttention_59820304499409
// MI455X (gfx1250) — matrix operand order repaired and checked by reading, not compiled, not run on hardware
//
#include <hip/hip_runtime.h>
#include <hip/hip_bf16.h>

typedef __attribute__((ext_vector_type(16))) _Float16 v16h;
typedef __attribute__((ext_vector_type(8)))  _Float16 v8h;
typedef __attribute__((ext_vector_type(4)))  _Float16 v4h;
typedef __attribute__((ext_vector_type(2)))  _Float16 v2h;
typedef __attribute__((ext_vector_type(8)))  float    v8f;
typedef __attribute__((ext_vector_type(4)))  float    v4f;

#define WMMA_F16(a,b,c) __builtin_amdgcn_wmma_f32_16x16x32_f16(false,(a),false,(b),(short)0,(c),false,false)

static constexpr int Bsz = 4, T = 2048, C = 768, H = 12, Dh = 64;
static constexpr int BT = Bsz * T;        // 8192
static constexpr int C3 = 3 * C;          // 2304

// ---- WMMA fragment loaders (ISA 7.12.2 lane layouts) -----------------------

// A-matrix 16x32 f16 fragment from row-major [rows x ld] f16 storage.
// lane L: row = row0 + (L&15); elem e -> k = k0 + 16*(L>>4) + e  (contiguous)
__device__ __forceinline__ v16h load_A_frag(const _Float16* base, int ld, int row0, int k0) {
  const int lane = threadIdx.x & 31;
  const int m = lane & 15, hf = lane >> 4;
  const _Float16* p = base + (size_t)(row0 + m) * ld + k0 + hf * 16;
  v8h lo = *(const v8h*)p;
  v8h hi = *(const v8h*)(p + 8);
  v16h a;
#pragma unroll
  for (int i = 0; i < 8; ++i) { a[i] = lo[i]; a[i + 8] = hi[i]; }
  return a;
}

// B-matrix 32x16 f16 fragment where B is stored transposed: bt[n][k], ld = K.
// lane L: col n = n0 + (L&15); elem e -> k = k0 + 16*(L>>4) + e  (contiguous)
__device__ __forceinline__ v16h load_B_fragT(const _Float16* bt, int ld, int n0, int k0) {
  const int lane = threadIdx.x & 31;
  const int n = lane & 15, hf = lane >> 4;
  return *(const v16h*)(bt + (size_t)(n0 + n) * ld + k0 + hf * 16);
}

// ---- pack kernels ----------------------------------------------------------

__global__ void cvt_f32_to_f16_v4(const v4f* __restrict__ s, v4h* __restrict__ d, int n4) {
  for (int i = blockIdx.x * blockDim.x + threadIdx.x; i < n4; i += gridDim.x * blockDim.x) {
    v4f x = s[i];
    v4h y;
#pragma unroll
    for (int j = 0; j < 4; ++j) y[j] = (_Float16)x[j];
    d[i] = y;
  }
}

// src [R x Cn] f32 row-major -> dst [Cn x R] f16 (i.e. W -> W^T)
__global__ void transpose_cvt(const float* __restrict__ s, _Float16* __restrict__ d, int R, int Cn) {
  int i = blockIdx.x * blockDim.x + threadIdx.x;
  if (i < R * Cn) {
    int r = i / Cn, c = i % Cn;
    d[(size_t)c * R + r] = (_Float16)s[i];
  }
}

// ---- QKV projection GEMM: 32x64 wave tile, double-buffered K loop ----------

__global__ void __launch_bounds__(256, 1)
qkv_gemm(const _Float16* __restrict__ xh, const _Float16* __restrict__ Wt,
         const float* __restrict__ bias, _Float16* __restrict__ Qh,
         _Float16* __restrict__ Kh, _Float16* __restrict__ Vt) {
  const int w = blockIdx.x * (blockDim.x >> 5) + (threadIdx.x >> 5);
  const int nt = w % (C3 / 64);              // 36 n-tiles of 64
  const int mt = w / (C3 / 64);              // 256 m-tiles of 32
  const int m0 = mt * 32, n0 = nt * 64;
  v8f acc[2][4] = {};

  v16h a0 = load_A_frag(xh, C, m0,      0);
  v16h a1 = load_A_frag(xh, C, m0 + 16, 0);
  v16h b0 = load_B_fragT(Wt, C, n0,      0);
  v16h b1 = load_B_fragT(Wt, C, n0 + 16, 0);
  v16h b2 = load_B_fragT(Wt, C, n0 + 32, 0);
  v16h b3 = load_B_fragT(Wt, C, n0 + 48, 0);

  for (int kn = 32; kn < C; kn += 32) {
    // prefetch next k-step while current WMMAs execute
    v16h na0 = load_A_frag(xh, C, m0,      kn);
    v16h na1 = load_A_frag(xh, C, m0 + 16, kn);
    v16h nb0 = load_B_fragT(Wt, C, n0,      kn);
    v16h nb1 = load_B_fragT(Wt, C, n0 + 16, kn);
    v16h nb2 = load_B_fragT(Wt, C, n0 + 32, kn);
    v16h nb3 = load_B_fragT(Wt, C, n0 + 48, kn);
    acc[0][0] = WMMA_F16(a0, b0, acc[0][0]);
    acc[0][1] = WMMA_F16(a0, b1, acc[0][1]);
    acc[0][2] = WMMA_F16(a0, b2, acc[0][2]);
    acc[0][3] = WMMA_F16(a0, b3, acc[0][3]);
    acc[1][0] = WMMA_F16(a1, b0, acc[1][0]);
    acc[1][1] = WMMA_F16(a1, b1, acc[1][1]);
    acc[1][2] = WMMA_F16(a1, b2, acc[1][2]);
    acc[1][3] = WMMA_F16(a1, b3, acc[1][3]);
    a0 = na0; a1 = na1; b0 = nb0; b1 = nb1; b2 = nb2; b3 = nb3;
  }
  acc[0][0] = WMMA_F16(a0, b0, acc[0][0]);
  acc[0][1] = WMMA_F16(a0, b1, acc[0][1]);
  acc[0][2] = WMMA_F16(a0, b2, acc[0][2]);
  acc[0][3] = WMMA_F16(a0, b3, acc[0][3]);
  acc[1][0] = WMMA_F16(a1, b0, acc[1][0]);
  acc[1][1] = WMMA_F16(a1, b1, acc[1][1]);
  acc[1][2] = WMMA_F16(a1, b2, acc[1][2]);
  acc[1][3] = WMMA_F16(a1, b3, acc[1][3]);

  const int lane = threadIdx.x & 31, hf = lane >> 4;
#pragma unroll
  for (int t = 0; t < 4; ++t) {
    const int n = n0 + 16 * t + (lane & 15);
    const float bv = bias[n];
#pragma unroll
    for (int rb = 0; rb < 2; ++rb) {
      // region branch is wave-uniform: the 64-wide n-stripe sits in one region
      if (n < C) {                            // Q, fold in 1/sqrt(D)
        const int h = n / Dh, d = n % Dh;
#pragma unroll
        for (int r = 0; r < 8; ++r) {
          const int m = m0 + rb * 16 + r + 8 * hf;
          const int bb = m >> 11, tok = m & (T - 1);
          Qh[(((size_t)(bb * H + h)) * T + tok) * Dh + d] =
              (_Float16)((acc[rb][t][r] + bv) * 0.125f);
        }
      } else if (n < 2 * C) {                 // K
        const int nn = n - C, h = nn / Dh, d = nn % Dh;
#pragma unroll
        for (int r = 0; r < 8; ++r) {
          const int m = m0 + rb * 16 + r + 8 * hf;
          const int bb = m >> 11, tok = m & (T - 1);
          Kh[(((size_t)(bb * H + h)) * T + tok) * Dh + d] = (_Float16)(acc[rb][t][r] + bv);
        }
      } else {                                // V^T [B,H,D,T]: 8 contiguous tokens
        const int nn = n - 2 * C, h = nn / Dh, d = nn % Dh;
        const int m = m0 + rb * 16 + 8 * hf;  // 8 consecutive tokens, same batch
        const int bb = m >> 11, tok = m & (T - 1);
        v8h pv;
#pragma unroll
        for (int r = 0; r < 8; ++r) pv[r] = (_Float16)(acc[rb][t][r] + bv);
        *(v8h*)&Vt[(((size_t)(bb * H + h)) * Dh + d) * T + tok] = pv;
      }
    }
  }
}

// ---- flash attention: one wave = one (b,h, 16-query tile) ------------------

__global__ void __launch_bounds__(256, 1)
attn(const _Float16* __restrict__ Qh, const _Float16* __restrict__ Kh,
     const _Float16* __restrict__ Vt, _Float16* __restrict__ yh) {
  const int w  = blockIdx.x * (blockDim.x >> 5) + (threadIdx.x >> 5);
  const int qt = w & 127;                     // 128 query tiles
  const int bh = w >> 7;                      // 48 (b,h) pairs
  const int b = bh / H, h = bh % H;
  const _Float16* Qp = Qh + (size_t)bh * T * Dh;   // [T x D]
  const _Float16* Kp = Kh + (size_t)bh * T * Dh;   // [T x D]
  const _Float16* Vp = Vt + (size_t)bh * Dh * T;   // [D x T] (V transposed)
  const int q0 = qt * 16;
  const int lane = threadIdx.x & 31, hf = lane >> 4;
  const int q = q0 + (lane & 15);             // this lane's query column

  // Q^T as B-fragments (reused all iterations): B[k=dim][n=query] = Qp[q][dim]
  const v16h bq0 = load_B_fragT(Qp, Dh, q0, 0);
  const v16h bq1 = load_B_fragT(Qp, Dh, q0, 32);

  v8f o0 = {}, o1 = {}, o2 = {}, o3 = {};     // O^T accumulators [64 dims x 16 q]
  float m_run = -1e30f, l_run = 0.0f;

  for (int kb = 0; kb <= q0; kb += 32) {      // two 16-key tiles per iteration
    // issue V^T loads early: they are consumed only after softmax VALU work,
    // so they overlap the score WMMAs + exp transcendentals
    v16h av0 = load_A_frag(Vp, T, 0,  kb);
    v16h av1 = load_A_frag(Vp, T, 16, kb);
    v16h av2 = load_A_frag(Vp, T, 32, kb);
    v16h av3 = load_A_frag(Vp, T, 48, kb);

    v8f s0 = {}, s1 = {};
    {                                         // S^T tile0 = K[kb..kb+15] @ Q^T
      v16h a = load_A_frag(Kp, Dh, kb, 0);  s0 = WMMA_F16(a, bq0, s0);
      a      = load_A_frag(Kp, Dh, kb, 32); s0 = WMMA_F16(a, bq1, s0);
    }
    const bool t1 = (kb + 31) < T;            // tile1 in bounds?
    if (t1) {
      v16h a = load_A_frag(Kp, Dh, kb + 16, 0);  s1 = WMMA_F16(a, bq0, s1);
      a      = load_A_frag(Kp, Dh, kb + 16, 32); s1 = WMMA_F16(a, bq1, s1);
    }
    // causal mask (selects, no EXEC change) + per-query tile max
    float tmax = -1e30f;
#pragma unroll
    for (int r = 0; r < 8; ++r) {
      const int k0i = kb + r + 8 * hf;
      s0[r] = (k0i <= q) ? s0[r] : -1e30f;
      s1[r] = (t1 && (k0i + 16) <= q) ? s1[r] : -1e30f;
      tmax = fmaxf(tmax, fmaxf(s0[r], s1[r]));
    }
    tmax = fmaxf(tmax, __shfl_xor(tmax, 16, 32));   // combine half-waves
    const float m_new = fmaxf(m_run, tmax);
    const float sc = __expf(m_run - m_new);
    float rsum = 0.0f;
    float p0[8], p1[8];
#pragma unroll
    for (int r = 0; r < 8; ++r) {
      p0[r] = __expf(s0[r] - m_new);
      p1[r] = __expf(s1[r] - m_new);
      rsum += p0[r] + p1[r];
    }
    rsum += __shfl_xor(rsum, 16, 32);
    l_run = l_run * sc + rsum;
    m_run = m_new;
#pragma unroll
    for (int r = 0; r < 8; ++r) { o0[r] *= sc; o1[r] *= sc; o2[r] *= sc; o3[r] *= sc; }

    // Repack P^T (D-layout) -> B-fragment of P: one half-wave exchange.
    // B elem e, lane L: P[key = kb + 16*(L>>4) + e][query = L&15]
    v16h pb;
#pragma unroll
    for (int r = 0; r < 8; ++r) {
      const float send = hf ? p0[r] : p1[r];
      const float recv = __shfl_xor(send, 16, 32);
      const float lo = hf ? recv : p0[r];
      const float hi = hf ? p1[r] : recv;
      pb[r]     = (_Float16)lo;
      pb[r + 8] = (_Float16)hi;
    }
    // O^T += V^T @ P   (V fragments already in flight / landed)
    o0 = WMMA_F16(av0, pb, o0);
    o1 = WMMA_F16(av1, pb, o1);
    o2 = WMMA_F16(av2, pb, o2);
    o3 = WMMA_F16(av3, pb, o3);
  }

  const float inv = 1.0f / l_run;
  _Float16* yrow = yh + ((size_t)(b * T + q)) * C + h * Dh;  // y[b][q][h*64+..]
#pragma unroll
  for (int r = 0; r < 8; r += 2) {            // pack pairs: dims d, d+1
    const int d = r + 8 * hf;
    v2h p;
    p[0] = (_Float16)(o0[r] * inv); p[1] = (_Float16)(o0[r + 1] * inv);
    *(v2h*)&yrow[d] = p;
    p[0] = (_Float16)(o1[r] * inv); p[1] = (_Float16)(o1[r + 1] * inv);
    *(v2h*)&yrow[16 + d] = p;
    p[0] = (_Float16)(o2[r] * inv); p[1] = (_Float16)(o2[r + 1] * inv);
    *(v2h*)&yrow[32 + d] = p;
    p[0] = (_Float16)(o3[r] * inv); p[1] = (_Float16)(o3[r + 1] * inv);
    *(v2h*)&yrow[48 + d] = p;
  }
}

// ---- output projection: 32x64 wave tile, double-buffered -------------------

__global__ void __launch_bounds__(256, 1)
proj_gemm(const _Float16* __restrict__ yh, const _Float16* __restrict__ Wt,
          const float* __restrict__ bias, float* __restrict__ out) {
  const int w = blockIdx.x * (blockDim.x >> 5) + (threadIdx.x >> 5);
  const int nt = w % (C / 64);               // 12
  const int mt = w / (C / 64);               // 256 m-tiles of 32
  const int m0 = mt * 32, n0 = nt * 64;
  v8f acc[2][4] = {};

  v16h a0 = load_A_frag(yh, C, m0,      0);
  v16h a1 = load_A_frag(yh, C, m0 + 16, 0);
  v16h b0 = load_B_fragT(Wt, C, n0,      0);
  v16h b1 = load_B_fragT(Wt, C, n0 + 16, 0);
  v16h b2 = load_B_fragT(Wt, C, n0 + 32, 0);
  v16h b3 = load_B_fragT(Wt, C, n0 + 48, 0);

  for (int kn = 32; kn < C; kn += 32) {
    v16h na0 = load_A_frag(yh, C, m0,      kn);
    v16h na1 = load_A_frag(yh, C, m0 + 16, kn);
    v16h nb0 = load_B_fragT(Wt, C, n0,      kn);
    v16h nb1 = load_B_fragT(Wt, C, n0 + 16, kn);
    v16h nb2 = load_B_fragT(Wt, C, n0 + 32, kn);
    v16h nb3 = load_B_fragT(Wt, C, n0 + 48, kn);
    acc[0][0] = WMMA_F16(a0, b0, acc[0][0]);
    acc[0][1] = WMMA_F16(a0, b1, acc[0][1]);
    acc[0][2] = WMMA_F16(a0, b2, acc[0][2]);
    acc[0][3] = WMMA_F16(a0, b3, acc[0][3]);
    acc[1][0] = WMMA_F16(a1, b0, acc[1][0]);
    acc[1][1] = WMMA_F16(a1, b1, acc[1][1]);
    acc[1][2] = WMMA_F16(a1, b2, acc[1][2]);
    acc[1][3] = WMMA_F16(a1, b3, acc[1][3]);
    a0 = na0; a1 = na1; b0 = nb0; b1 = nb1; b2 = nb2; b3 = nb3;
  }
  acc[0][0] = WMMA_F16(a0, b0, acc[0][0]);
  acc[0][1] = WMMA_F16(a0, b1, acc[0][1]);
  acc[0][2] = WMMA_F16(a0, b2, acc[0][2]);
  acc[0][3] = WMMA_F16(a0, b3, acc[0][3]);
  acc[1][0] = WMMA_F16(a1, b0, acc[1][0]);
  acc[1][1] = WMMA_F16(a1, b1, acc[1][1]);
  acc[1][2] = WMMA_F16(a1, b2, acc[1][2]);
  acc[1][3] = WMMA_F16(a1, b3, acc[1][3]);

  const int lane = threadIdx.x & 31, hf = lane >> 4;
#pragma unroll
  for (int t = 0; t < 4; ++t) {
    const int n = n0 + 16 * t + (lane & 15);
    const float bv = bias[n];
#pragma unroll
    for (int rb = 0; rb < 2; ++rb)
#pragma unroll
      for (int r = 0; r < 8; ++r) {
        const int m = m0 + rb * 16 + r + 8 * hf;
        out[(size_t)m * C + n] = acc[rb][t][r] + bv;
      }
  }
}

// ---- launcher --------------------------------------------------------------

extern "C" void kernel_launch(void* const* d_in, const int* in_sizes, int n_in,
                              void* d_out, int out_size, void* d_ws, size_t ws_size,
                              hipStream_t stream) {
  const float* x     = (const float*)d_in[0];
  const float* Wqkv  = (const float*)d_in[1];
  const float* bqkv  = (const float*)d_in[2];
  const float* Wproj = (const float*)d_in[3];
  const float* bproj = (const float*)d_in[4];
  float* out = (float*)d_out;

  char* ws = (char*)d_ws;
  size_t off = 0;
  auto alloc = [&](size_t bytes) -> char* {
    char* p = ws + off;
    off += (bytes + 255) & ~(size_t)255;
    return p;
  };
  _Float16* xh     = (_Float16*)alloc((size_t)BT * C * 2);           // x in f16
  _Float16* WqkvT  = (_Float16*)alloc((size_t)C3 * C * 2);           // Wqkv^T f16
  _Float16* WprojT = (_Float16*)alloc((size_t)C * C * 2);            // Wproj^T f16
  _Float16* Qh     = (_Float16*)alloc((size_t)Bsz * H * T * Dh * 2); // [B,H,T,D]
  _Float16* Kh     = (_Float16*)alloc((size_t)Bsz * H * T * Dh * 2); // [B,H,T,D]
  _Float16* Vt     = (_Float16*)alloc((size_t)Bsz * H * Dh * T * 2); // [B,H,D,T]
  _Float16* yh     = (_Float16*)alloc((size_t)BT * C * 2);           // y in f16
  (void)ws_size;

  cvt_f32_to_f16_v4<<<3072, 256, 0, stream>>>((const v4f*)x, (v4h*)xh, BT * C / 4);
  transpose_cvt<<<(C * C3 + 255) / 256, 256, 0, stream>>>(Wqkv, WqkvT, C, C3);
  transpose_cvt<<<(C * C + 255) / 256, 256, 0, stream>>>(Wproj, WprojT, C, C);

  qkv_gemm<<<(BT / 32) * (C3 / 64) / 8, 256, 0, stream>>>(xh, WqkvT, bqkv, Qh, Kh, Vt);
  attn<<<(Bsz * H * (T / 16)) / 8, 256, 0, stream>>>(Qh, Kh, Vt, yh);
  proj_gemm<<<(BT / 32) * (C / 64) / 8, 256, 0, stream>>>(yh, WprojT, bproj, out);
}
